// GLKSDecoder_40707700031898
// MI455X (gfx1250) — matrix operand order repaired and checked by reading, not compiled, not run on hardware
//
#include <hip/hip_runtime.h>
#include <hip/hip_bf16.h>

// ---------------------------------------------------------------------------
// GLKS decoder for MI455X (gfx1250, wave32, WMMA).
// Dominant GEMMs run on v_wmma_f32_16x16x32_f16 with f32 accumulation.
// Weights are pre-converted once into a "fragment-swizzled" f16 layout so each
// lane's 16-half B fragment is a single contiguous 32-byte load.
// Workspace requirement: ~136 MB.
// ---------------------------------------------------------------------------

typedef __attribute__((ext_vector_type(16))) _Float16 v16h;
typedef __attribute__((ext_vector_type(8)))  _Float16 v8h;
typedef __attribute__((ext_vector_type(8)))  float    v8f;

enum : int { Bb = 32, Tt = 128, Ee = 256, Hh = 512, Aa = 256, Vv = 32000,
             Ss = 64, OUTIN_D = 2304 };
#define NEGV (-1.0e9f)

// ---------------------------------------------------------------------------
// B-fragment swizzle for v_wmma_f32_16x16x32_f16:
//   dst[(((n0*Ks + ks)*32 + lane)*16 + j] = (f16) src[k][n]
//   with k = ks*32 + (j>>3)*16 + (lane>>4)*8 + (j&7), n = n0*16 + (lane&15)
// (lane<16 holds K=0..7,16..23 of column n, lane>=16 holds K=8..15,24..31)
// ---------------------------------------------------------------------------
__global__ void swizzle_b(_Float16* __restrict__ dst, const float* __restrict__ src,
                          int K, int N) {
  long idx = (long)blockIdx.x * 256 + threadIdx.x;
  long total = (long)K * N;
  if (idx >= total) return;
  int j    = (int)(idx & 15);
  int lane = (int)((idx >> 4) & 31);
  long rest = idx >> 9;
  int Ks = K >> 5;
  int ks = (int)(rest % Ks);
  int n0 = (int)(rest / Ks);
  int k = ks * 32 + (j >> 3) * 16 + (lane >> 4) * 8 + (j & 7);
  int n = n0 * 16 + (lane & 15);
  dst[idx] = (_Float16)src[(size_t)k * N + n];
}

// ---------------------------------------------------------------------------
// Generic WMMA GEMM: C[M,N] = A[M,K](f32,row-major) * Bswz(f16) (+ bias[N])
// One wave per 16x16 tile; 4 waves per block cover 4 consecutive N tiles.
// Requires M%16==0, N%64==0, K%32==0.
// ---------------------------------------------------------------------------
__global__ __launch_bounds__(128) void gemm_ws(const float* __restrict__ A,
                                               const _Float16* __restrict__ Bswz,
                                               const float* __restrict__ bias,
                                               float* __restrict__ C,
                                               int M, int N, int K) {
  int tid = threadIdx.x, wave = tid >> 5, lane = tid & 31;
  int n4 = N >> 6;                       // number of 4-tile groups in N
  int mt = blockIdx.x / n4;
  int nt = (blockIdx.x % n4) * 4 + wave;
  int Ks = K >> 5;
  int m = lane & 15, hlf = lane >> 4;
  const float* arow = A + (size_t)(mt * 16 + m) * K;
  v8f c = {};
  for (int ks = 0; ks < Ks; ++ks) {
    const float* p = arow + ks * 32;
    v16h a;
#pragma unroll
    for (int j = 0; j < 8; ++j) a[j]     = (_Float16)p[hlf * 8 + j];
#pragma unroll
    for (int j = 0; j < 8; ++j) a[8 + j] = (_Float16)p[16 + hlf * 8 + j];
    v16h b = *(const v16h*)(Bswz + ((((size_t)nt * Ks + ks) * 32 + lane) << 4));
    c = __builtin_amdgcn_wmma_f32_16x16x32_f16(false, a, false, b, (short)0, c,
                                               false, false);
  }
  int col = nt * 16 + m;
  float bv = bias ? bias[col] : 0.f;
  int rowb = mt * 16 + hlf * 8;
#pragma unroll
  for (int r = 0; r < 8; ++r) C[(size_t)(rowb + r) * N + col] = c[r] + bv;
}

// ---------------------------------------------------------------------------
// Prep kernels
// ---------------------------------------------------------------------------
__global__ void gather_x(float* __restrict__ x_all, const float* __restrict__ emb,
                         const int* __restrict__ tokens) {
  long idx = (long)blockIdx.x * 256 + threadIdx.x;
  if (idx >= (long)Bb * Tt * Ee) return;
  int row = (int)(idx >> 8), e = (int)(idx & 255);
  x_all[idx] = emb[(size_t)tokens[row] * Ee + e];
}

// Wpre (256 x 2048) = [w_ih | src_wq rows 0..255 | cue_wq rows 0..255]
__global__ void pack_pre(float* __restrict__ W, const float* __restrict__ w_ih,
                         const float* __restrict__ swq, const float* __restrict__ cwq) {
  long idx = (long)blockIdx.x * 256 + threadIdx.x;
  if (idx >= (long)Ee * 2048) return;
  int k = (int)(idx / 2048), c = (int)(idx % 2048);
  float v;
  if (c < 1536)      v = w_ih[(size_t)k * 1536 + c];
  else if (c < 1792) v = swq[(size_t)k * Aa + (c - 1536)];
  else               v = cwq[(size_t)k * Aa + (c - 1792)];
  W[idx] = v;
}

// Wstep (512 x 2048) = [w_hh | src_wq rows 256..767 | cue_wq rows 256..767]
__global__ void pack_step(float* __restrict__ W, const float* __restrict__ w_hh,
                          const float* __restrict__ swq, const float* __restrict__ cwq) {
  long idx = (long)blockIdx.x * 256 + threadIdx.x;
  if (idx >= (long)Hh * 2048) return;
  int k = (int)(idx / 2048), c = (int)(idx % 2048);
  float v;
  if (c < 1536)      v = w_hh[(size_t)k * 1536 + c];
  else if (c < 1792) v = swq[(size_t)(Ee + k) * Aa + (c - 1536)];
  else               v = cwq[(size_t)(Ee + k) * Aa + (c - 1792)];
  W[idx] = v;
}

// Wkq (512 x 512) = [src_wq rows 768..1279 | cue_wq rows 768..1279]
__global__ void pack_kq(float* __restrict__ W, const float* __restrict__ swq,
                        const float* __restrict__ cwq) {
  long idx = (long)blockIdx.x * 256 + threadIdx.x;
  if (idx >= (long)Hh * 512) return;
  int k = (int)(idx / 512), c = (int)(idx % 512);
  W[idx] = (c < 256) ? swq[(size_t)(Ee + Hh + k) * Aa + c]
                     : cwq[(size_t)(Ee + Hh + k) * Aa + (c - 256)];
}

__global__ void pack_biases(float* __restrict__ bpre, float* __restrict__ bstep,
                            float* __restrict__ bkq, const float* __restrict__ b_ih,
                            const float* __restrict__ b_hh,
                            const float* __restrict__ sbq,
                            const float* __restrict__ cbq) {
  int idx = blockIdx.x * 256 + threadIdx.x;
  if (idx < 2048) {
    bpre[idx]  = (idx < 1536) ? b_ih[idx] : 0.f;
    bstep[idx] = (idx < 1536) ? b_hh[idx] : 0.f;
    if (idx < 512) bkq[idx] = (idx < 256) ? sbq[idx] : cbq[idx - 256];
  }
}

__global__ void init_h(float* __restrict__ h, const float* __restrict__ h0) {
  int i = blockIdx.x * 256 + threadIdx.x;
  if (i < Bb * Hh) h[i] = h0[i];
}

__global__ void copy_hfinal(float* __restrict__ dst, const float* __restrict__ h) {
  int i = blockIdx.x * 256 + threadIdx.x;
  if (i < Bb * Hh) dst[i] = h[i];
}

// ---------------------------------------------------------------------------
// Per-step elementwise: GRU gates + two tanh-MLP attentions + assembly.
// One block (256 threads) per batch element.
// ---------------------------------------------------------------------------
__device__ void mlp_attn_ctx(const float* __restrict__ qact,
                             const float* __restrict__ mk,
                             const float* __restrict__ mem,
                             const float* __restrict__ vvec, int slen, bool valid,
                             float* __restrict__ outrow, float* sc, float* red,
                             float* part, int tid, int b) {
  int s = tid >> 2, p = tid & 3;
  const float* mkrow = mk + (size_t)(b * Ss + s) * Aa;
  float acc = 0.f;
  for (int a = p * 64; a < p * 64 + 64; ++a)
    acc += tanhf(qact[a] + mkrow[a]) * vvec[a];
  part[tid] = acc;
  __syncthreads();
  if (tid < Ss) {
    float t = part[tid * 4] + part[tid * 4 + 1] + part[tid * 4 + 2] + part[tid * 4 + 3];
    t = (tid < slen) ? t : NEGV;
    sc[tid] = t;
    red[tid] = t;
  }
  __syncthreads();
  for (int off = 32; off > 0; off >>= 1) {
    if (tid < off) red[tid] = fmaxf(red[tid], red[tid + off]);
    __syncthreads();
  }
  float mx = red[0];
  __syncthreads();
  if (tid < Ss) { float e = expf(sc[tid] - mx); sc[tid] = e; red[tid] = e; }
  __syncthreads();
  for (int off = 32; off > 0; off >>= 1) {
    if (tid < off) red[tid] += red[tid + off];
    __syncthreads();
  }
  float inv = 1.f / red[0];
  __syncthreads();
  const float* mb = mem + (size_t)b * Ss * Hh;
  for (int hc = tid; hc < Hh; hc += 256) {
    float c = 0.f;
    for (int s2 = 0; s2 < Ss; ++s2) c += sc[s2] * mb[(size_t)s2 * Hh + hc];
    outrow[hc] = valid ? c * inv : 0.f;
  }
  __syncthreads();
}

__global__ __launch_bounds__(256) void step_update(
    int t, const float* __restrict__ x_all, const float* __restrict__ giqx,
    const float* __restrict__ cstep, float* __restrict__ h,
    const float* __restrict__ knowledge, const float* __restrict__ kq,
    const float* __restrict__ smk, const float* __restrict__ cmk,
    const float* __restrict__ smem, const float* __restrict__ cmem,
    const float* __restrict__ sv, const float* __restrict__ cv,
    const int* __restrict__ lengths, const int* __restrict__ slen,
    const int* __restrict__ clen, float* __restrict__ outin,
    float* __restrict__ hidout) {
  int b = blockIdx.x, tid = threadIdx.x;
  __shared__ float qs[Aa], qc[Aa], sc[Ss], red[Ss], part[256];
  int row = b * Tt + t;
  bool valid = t < lengths[b];
  const float* gi = giqx + (size_t)row * 2048;
  const float* gh = cstep + (size_t)b * 2048;
  float* orow = outin + (size_t)row * OUTIN_D;

  // attention query activations (x-part + h-part + knowledge-part + biases)
  qs[tid] = gi[1536 + tid] + gh[1536 + tid] + kq[b * 512 + tid];
  qc[tid] = gi[1792 + tid] + gh[1792 + tid] + kq[b * 512 + 256 + tid];

  // out_in cols 0..255 (x) and 768..1279 (knowledge)
  orow[tid] = valid ? x_all[(size_t)row * Ee + tid] : 0.f;
  for (int j = tid; j < Hh; j += 256)
    orow[768 + j] = valid ? knowledge[b * Hh + j] : 0.f;

  // GRU cell
  for (int j = tid; j < Hh; j += 256) {
    float ir = gi[j], iz = gi[512 + j], in_ = gi[1024 + j];
    float hr = gh[j], hz = gh[512 + j], hn = gh[1024 + j];
    float r = 1.f / (1.f + expf(-(ir + hr)));
    float z = 1.f / (1.f + expf(-(iz + hz)));
    float n = tanhf(in_ + r * hn);
    float hp = h[b * Hh + j];
    float hnew = (1.f - z) * n + z * hp;
    orow[256 + j] = valid ? hnew : 0.f;
    hidout[(size_t)row * Hh + j] = valid ? hnew : 0.f;
    h[b * Hh + j] = valid ? hnew : hp;
  }
  __syncthreads();
  mlp_attn_ctx(qs, smk, smem, sv, slen[b], valid, orow + 1280, sc, red, part, tid, b);
  mlp_attn_ctx(qc, cmk, cmem, cv, clen[b], valid, orow + 1792, sc, red, part, tid, b);
}

// ---------------------------------------------------------------------------
// Logits band kernel: 16-row band of out1 staged to LDS (f16), 8 waves stream
// N tiles of swizzled w_out2, write logits to d_out, keep online log-sum-exp.
// ---------------------------------------------------------------------------
#define ASTR 520  // padded LDS row stride (halves) to avoid bank conflicts

__global__ __launch_bounds__(256) void logits_band(
    const float* __restrict__ out1, const _Float16* __restrict__ w2s,
    const float* __restrict__ b2, float* __restrict__ dlog,
    float* __restrict__ lse) {
  __shared__ __align__(16) _Float16 Asm[16 * ASTR];
  __shared__ float redM[8 * 32 * 8];
  __shared__ float redS[8 * 32 * 8];
  int tid = threadIdx.x, wave = tid >> 5, lane = tid & 31;
  int m0 = blockIdx.x * 16;

  for (int i = tid; i < 16 * Hh; i += 256) {
    int rr = i >> 9, kk = i & 511;
    Asm[rr * ASTR + kk] = (_Float16)out1[(size_t)(m0 + rr) * Hh + kk];
  }
  __syncthreads();

  int m = lane & 15, hlf = lane >> 4;
  float mx[8], sm[8];
#pragma unroll
  for (int r = 0; r < 8; ++r) { mx[r] = -3.0e38f; sm[r] = 0.f; }

  const int NT = Vv / 16;  // 2000
  for (int nt = wave; nt < NT; nt += 8) {
    v8f c = {};
#pragma unroll
    for (int ks = 0; ks < 16; ++ks) {
      const _Float16* pa = &Asm[m * ASTR + ks * 32 + hlf * 8];
      v8h lo = *(const v8h*)pa;
      v8h hi = *(const v8h*)(pa + 16);
      v16h a;
#pragma unroll
      for (int j = 0; j < 8; ++j) { a[j] = lo[j]; a[8 + j] = hi[j]; }
      v16h b = *(const v16h*)(w2s + ((((size_t)nt * 16 + ks) * 32 + lane) << 4));
      c = __builtin_amdgcn_wmma_f32_16x16x32_f16(false, a, false, b, (short)0, c,
                                                 false, false);
    }
    int col = nt * 16 + m;
    float bias = b2[col];
    size_t base = (size_t)(m0 + hlf * 8) * Vv + col;
#pragma unroll
    for (int r = 0; r < 8; ++r) {
      float val = c[r] + bias;
      dlog[base + (size_t)r * Vv] = val;
      float nm = fmaxf(mx[r], val);
      sm[r] = sm[r] * expf(mx[r] - nm) + expf(val - nm);
      mx[r] = nm;
    }
  }
#pragma unroll
  for (int r = 0; r < 8; ++r) {
    redM[(wave * 32 + lane) * 8 + r] = mx[r];
    redS[(wave * 32 + lane) * 8 + r] = sm[r];
  }
  __syncthreads();
  if (tid < 16) {
    int rowl = tid, l0 = (rowl < 8) ? 0 : 16, slot = rowl & 7;
    float M = -3.0e38f, S = 0.f;
    for (int w = 0; w < 8; ++w)
      for (int li = 0; li < 16; ++li) {
        int idx = (w * 32 + l0 + li) * 8 + slot;
        float M2 = redM[idx], S2 = redS[idx];
        float nm = fmaxf(M, M2);
        S = S * expf(M - nm) + S2 * expf(M2 - nm);
        M = nm;
      }
    lse[m0 + rowl] = M + logf(S);
  }
}

__global__ void finalize_logsoftmax(float* __restrict__ dlog,
                                    const float* __restrict__ lse) {
  size_t idx = (size_t)blockIdx.x * 256 + threadIdx.x;
  size_t total = (size_t)Bb * Tt * Vv;
  if (idx >= total) return;
  int row = (int)(idx / Vv);
  dlog[idx] -= lse[row];
}

// ---------------------------------------------------------------------------
// Host side
// ---------------------------------------------------------------------------
static inline size_t alignup(size_t x) { return (x + 255) & ~(size_t)255; }

extern "C" void kernel_launch(void* const* d_in, const int* in_sizes, int n_in,
                              void* d_out, int out_size, void* d_ws, size_t ws_size,
                              hipStream_t stream) {
  const int*   tokens      = (const int*)d_in[0];
  const int*   lengths     = (const int*)d_in[1];
  const float* hidden0     = (const float*)d_in[2];
  const float* knowledge   = (const float*)d_in[3];
  const float* src_mem     = (const float*)d_in[4];
  const int*   src_lengths = (const int*)d_in[5];
  const float* cue_mem     = (const float*)d_in[6];
  const int*   cue_lengths = (const int*)d_in[7];
  const float* emb         = (const float*)d_in[8];
  const float* w_ih        = (const float*)d_in[9];
  const float* w_hh        = (const float*)d_in[10];
  const float* b_ih        = (const float*)d_in[11];
  const float* b_hh        = (const float*)d_in[12];
  const float* src_wq      = (const float*)d_in[13];
  const float* src_bq      = (const float*)d_in[14];
  const float* src_wm      = (const float*)d_in[15];
  const float* src_v       = (const float*)d_in[16];
  const float* cue_wq      = (const float*)d_in[17];
  const float* cue_bq      = (const float*)d_in[18];
  const float* cue_wm      = (const float*)d_in[19];
  const float* cue_v       = (const float*)d_in[20];
  const float* w_out1      = (const float*)d_in[21];
  const float* b_out1      = (const float*)d_in[22];
  const float* w_out2      = (const float*)d_in[23];
  const float* b_out2      = (const float*)d_in[24];

  char* ws = (char*)d_ws;
  size_t off = 0;
  auto alloc = [&](size_t bytes) { char* p = ws + off; off += alignup(bytes); return p; };

  float*    X_ALL  = (float*)alloc((size_t)Bb * Tt * Ee * 4);
  float*    GIQX   = (float*)alloc((size_t)Bb * Tt * 2048 * 4);
  float*    WPRE32 = (float*)alloc((size_t)Ee * 2048 * 4);
  float*    WSTP32 = (float*)alloc((size_t)Hh * 2048 * 4);
  float*    WKQ32  = (float*)alloc((size_t)Hh * 512 * 4);
  _Float16* WPREH  = (_Float16*)alloc((size_t)Ee * 2048 * 2);
  _Float16* WSTPH  = (_Float16*)alloc((size_t)Hh * 2048 * 2);
  _Float16* WKQH   = (_Float16*)alloc((size_t)Hh * 512 * 2);
  _Float16* WO1H   = (_Float16*)alloc((size_t)OUTIN_D * Hh * 2);
  _Float16* WO2H   = (_Float16*)alloc((size_t)Hh * Vv * 2);
  _Float16* SWMH   = (_Float16*)alloc((size_t)Hh * Aa * 2);
  _Float16* CWMH   = (_Float16*)alloc((size_t)Hh * Aa * 2);
  float*    BPRE   = (float*)alloc(2048 * 4);
  float*    BSTEP  = (float*)alloc(2048 * 4);
  float*    BKQ    = (float*)alloc(512 * 4);
  float*    KQ     = (float*)alloc((size_t)Bb * 512 * 4);
  float*    SMK    = (float*)alloc((size_t)Bb * Ss * Aa * 4);
  float*    CMK    = (float*)alloc((size_t)Bb * Ss * Aa * 4);
  float*    CSTEP  = (float*)alloc((size_t)Bb * 2048 * 4);
  float*    HBUF   = (float*)alloc((size_t)Bb * Hh * 4);
  float*    OUTIN  = (float*)alloc((size_t)Bb * Tt * OUTIN_D * 4);
  float*    OUT1   = (float*)alloc((size_t)Bb * Tt * Hh * 4);
  float*    LSE    = (float*)alloc((size_t)Bb * Tt * 4);
  (void)ws_size;

  float* out_log = (float*)d_out;
  float* out_hid = out_log + (size_t)Bb * Tt * Vv;
  float* out_hf  = out_hid + (size_t)Bb * Tt * Hh;

  auto blk = [](long n) { return (unsigned)((n + 255) / 256); };

  // --- prep ---
  gather_x<<<blk((long)Bb * Tt * Ee), 256, 0, stream>>>(X_ALL, emb, tokens);
  pack_pre<<<blk((long)Ee * 2048), 256, 0, stream>>>(WPRE32, w_ih, src_wq, cue_wq);
  pack_step<<<blk((long)Hh * 2048), 256, 0, stream>>>(WSTP32, w_hh, src_wq, cue_wq);
  pack_kq<<<blk((long)Hh * 512), 256, 0, stream>>>(WKQ32, src_wq, cue_wq);
  pack_biases<<<8, 256, 0, stream>>>(BPRE, BSTEP, BKQ, b_ih, b_hh, src_bq, cue_bq);

  swizzle_b<<<blk((long)Ee * 2048), 256, 0, stream>>>(WPREH, WPRE32, Ee, 2048);
  swizzle_b<<<blk((long)Hh * 2048), 256, 0, stream>>>(WSTPH, WSTP32, Hh, 2048);
  swizzle_b<<<blk((long)Hh * 512), 256, 0, stream>>>(WKQH, WKQ32, Hh, 512);
  swizzle_b<<<blk((long)OUTIN_D * Hh), 256, 0, stream>>>(WO1H, w_out1, OUTIN_D, Hh);
  swizzle_b<<<blk((long)Hh * Vv), 256, 0, stream>>>(WO2H, w_out2, Hh, Vv);
  swizzle_b<<<blk((long)Hh * Aa), 256, 0, stream>>>(SWMH, src_wm, Hh, Aa);
  swizzle_b<<<blk((long)Hh * Aa), 256, 0, stream>>>(CWMH, cue_wm, Hh, Aa);

  // giqx = x_all @ [w_ih | wq_x] + [b_ih | 0]   (M=4096, N=2048, K=256)
  gemm_ws<<<(4096 / 16) * (2048 / 64), 128, 0, stream>>>(X_ALL, WPREH, BPRE, GIQX,
                                                         4096, 2048, 256);
  // kq = knowledge @ [wq_k] + [bq]              (M=32, N=512, K=512)
  gemm_ws<<<(32 / 16) * (512 / 64), 128, 0, stream>>>(knowledge, WKQH, BKQ, KQ,
                                                      32, 512, 512);
  // attention memory keys                       (M=2048, N=256, K=512)
  gemm_ws<<<(2048 / 16) * (256 / 64), 128, 0, stream>>>(src_mem, SWMH, nullptr, SMK,
                                                        2048, 256, 512);
  gemm_ws<<<(2048 / 16) * (256 / 64), 128, 0, stream>>>(cue_mem, CWMH, nullptr, CMK,
                                                        2048, 256, 512);
  init_h<<<blk(Bb * Hh), 256, 0, stream>>>(HBUF, hidden0);

  // --- sequential GRU + attention loop ---
  for (int t = 0; t < Tt; ++t) {
    // h @ [w_hh | wq_h] + [b_hh | 0]            (M=32, N=2048, K=512)
    gemm_ws<<<(32 / 16) * (2048 / 64), 128, 0, stream>>>(HBUF, WSTPH, BSTEP, CSTEP,
                                                         32, 2048, 512);
    step_update<<<Bb, 256, 0, stream>>>(t, X_ALL, GIQX, CSTEP, HBUF, knowledge, KQ,
                                        SMK, CMK, src_mem, cue_mem, src_v, cue_v,
                                        lengths, src_lengths, cue_lengths, OUTIN,
                                        out_hid);
  }
  copy_hfinal<<<blk(Bb * Hh), 256, 0, stream>>>(out_hf, HBUF);

  // --- output projections ---
  gemm_ws<<<(4096 / 16) * (512 / 64), 128, 0, stream>>>(OUTIN, WO1H, b_out1, OUT1,
                                                        4096, 512, OUTIN_D);
  logits_band<<<4096 / 16, 256, 0, stream>>>(OUT1, WO2H, b_out2, out_log, LSE);
  finalize_logsoftmax<<<blk((long)Bb * Tt * Vv), 256, 0, stream>>>(out_log, LSE);
  (void)in_sizes; (void)n_in; (void)out_size;
}
